// RetrievalLayer_64261300683311
// MI455X (gfx1250) — matrix operand order repaired and checked by reading, not compiled, not run on hardware
//
#include <hip/hip_runtime.h>
#include <hip/hip_bf16.h>
#include <math.h>

typedef __attribute__((ext_vector_type(16))) _Float16 v16h;
typedef __attribute__((ext_vector_type(8)))  float    v8f;
typedef __attribute__((ext_vector_type(4)))  _Float16 h4;

#define NN 2
#define CC 8192
#define HID 4096
#define RDIM 512
#define NH 8
#define PH 64          // per-head dim
#define DCH 128        // landmark chunks
#define TOPK 16
#define ROWS (NN*CC)   // 16384 flattened (n,c) rows
#define TK 64          // K-step of the projection GEMM

__device__ __forceinline__ float neg_inf() { return -__builtin_inff(); }

// CDNA5 async global->LDS copy (ASYNCcnt-tracked DMA-style staging).
// LDS byte address = low 32 bits of the generic pointer (ISA §10.2 flat-LDS).
__device__ __forceinline__ void async_copy_b128(void* lds, const void* g) {
    unsigned l = (unsigned)(uintptr_t)lds;
    asm volatile("global_load_async_to_lds_b128 %0, %1, off"
                 :: "v"(l), "v"(g) : "memory");
}
__device__ __forceinline__ void wait_async_le8() {
    asm volatile("s_wait_asynccnt 0x8" ::: "memory");
}
__device__ __forceinline__ void wait_async_0() {
    asm volatile("s_wait_asynccnt 0x0" ::: "memory");
}

// 16-bit A-matrix 16x32 fragment K offset (ISA §7.12.2)
__device__ __forceinline__ int kA(int v, int half) {
    return (v < 4) ? (2 * v + 8 * half) : (16 + 2 * (v - 4) + 8 * half);
}

// ---- Kernel 0: fused RMSNorm + pre_norm_weight, fp32 -> f16 (one wave/row) --
__global__ void __launch_bounds__(256)
rms_norm_f16_kernel(const float* __restrict__ hs, const float* __restrict__ pnw,
                    _Float16* __restrict__ xh) {
    const int row  = blockIdx.x * 8 + (threadIdx.x >> 5);
    const int lane = threadIdx.x & 31;
    const float* p = hs + (size_t)row * HID;
    float ss = 0.0f;
    #pragma unroll 4
    for (int k = lane * 4; k < HID; k += 128) {
        __builtin_prefetch(p + k + 1024, 0, 3);   // global_prefetch_b8
        float4 v = *(const float4*)(p + k);
        ss += v.x * v.x + v.y * v.y + v.z * v.z + v.w * v.w;
    }
    #pragma unroll
    for (int off = 16; off; off >>= 1) ss += __shfl_xor(ss, off, 32);
    const float scale = rsqrtf(ss * (1.0f / (float)HID) + 1e-6f);
    _Float16* o = xh + (size_t)row * HID;
    #pragma unroll 4
    for (int k = lane * 4; k < HID; k += 128) {
        float4 v = *(const float4*)(p + k);
        float4 w = *(const float4*)(pnw + k);
        h4 r = { (_Float16)(v.x * scale * w.x), (_Float16)(v.y * scale * w.y),
                 (_Float16)(v.z * scale * w.z), (_Float16)(v.w * scale * w.w) };
        *(h4*)(o + k) = r;
    }
}

// ---- Kernel 0b: plain fp32 -> f16 (x mul), one float4 per thread -----------
__global__ void __launch_bounds__(256)
f32_to_f16_kernel(const float* __restrict__ src, _Float16* __restrict__ dst,
                  float mul) {
    const int i = blockIdx.x * blockDim.x + threadIdx.x;
    float4 v = ((const float4*)src)[i];
    h4 r = { (_Float16)(v.x * mul), (_Float16)(v.y * mul),
             (_Float16)(v.z * mul), (_Float16)(v.w * mul) };
    ((h4*)dst)[i] = r;
}

// ---- Kernel 1: projection GEMM, 128x128 tile, async double-buffered --------
// qh[m][n] = f16( sum_k xh[m][k] * wf[n][k] ),  M=16384 K=4096 N=512
__global__ void __launch_bounds__(256)
proj_gemm_kernel(const _Float16* __restrict__ xh, const _Float16* __restrict__ wf,
                 _Float16* __restrict__ qh) {
    __shared__ _Float16 Ab[2][128][72];   // [buf][m][k], 144B row stride (16B-aligned)
    __shared__ _Float16 Bb[2][128][72];   // [buf][n][k]

    const int t       = threadIdx.x;
    const int rowBase = blockIdx.x * 128;
    const int colBase = blockIdx.y * 128;

    auto stage = [&](int buf, int kt) {   // 8 async ops per thread
        #pragma unroll
        for (int i = 0; i < 4; ++i) {
            int e  = t + i * 256;         // 0..1023 chunks of 8 halves
            int m  = e >> 3;
            int kc = (e & 7) * 8;
            async_copy_b128(&Ab[buf][m][kc],
                            xh + (size_t)(rowBase + m) * HID + kt + kc);
            async_copy_b128(&Bb[buf][m][kc],
                            wf + (size_t)(colBase + m) * HID + kt + kc);
        }
    };

    const int lane = t & 31;
    const int wv   = t >> 5;
    const int half = lane >> 4;
    const int l16  = lane & 15;
    const int rb0  = (wv & 3) * 2;        // two 16-row bands per wave
    const int cb0  = (wv >> 2) * 4;       // four 16-col tiles per wave

    v8f acc[2][4] = {};

    stage(0, 0);
    for (int it = 0; it < HID / TK; ++it) {
        const int buf = it & 1;
        if (it + 1 < HID / TK) {
            stage(buf ^ 1, (it + 1) * TK);
            wait_async_le8();             // current buf done; prefetch in flight
        } else {
            wait_async_0();
        }
        __syncthreads();

        #pragma unroll
        for (int ks = 0; ks < TK; ks += 32) {
            v16h a0, a1;
            #pragma unroll
            for (int v = 0; v < 8; ++v) {
                int kb = ks + kA(v, half);
                a0[2 * v]     = Ab[buf][rb0 * 16 + l16][kb];
                a0[2 * v + 1] = Ab[buf][rb0 * 16 + l16][kb + 1];
                a1[2 * v]     = Ab[buf][(rb0 + 1) * 16 + l16][kb];
                a1[2 * v + 1] = Ab[buf][(rb0 + 1) * 16 + l16][kb + 1];
            }
            #pragma unroll
            for (int cb = 0; cb < 4; ++cb) {
                v16h b;
                #pragma unroll
                for (int r = 0; r < 8; ++r) {
                    int kb = ks + kA(r, half);
                    b[2 * r]     = Bb[buf][(cb0 + cb) * 16 + l16][kb];
                    b[2 * r + 1] = Bb[buf][(cb0 + cb) * 16 + l16][kb + 1];
                }
                acc[0][cb] = __builtin_amdgcn_wmma_f32_16x16x32_f16(
                    false, a0, false, b, (short)0, acc[0][cb], false, false);
                acc[1][cb] = __builtin_amdgcn_wmma_f32_16x16x32_f16(
                    false, a1, false, b, (short)0, acc[1][cb], false, false);
            }
        }
        __syncthreads();
    }

    // epilogue: C layout — VGPR i: lanes0-15 M=i, lanes16-31 M=8+i
    #pragma unroll
    for (int rr = 0; rr < 2; ++rr) {
        const int mrow = rowBase + (rb0 + rr) * 16 + half * 8;
        #pragma unroll
        for (int cb = 0; cb < 4; ++cb) {
            const int ncol = colBase + (cb0 + cb) * 16 + l16;
            #pragma unroll
            for (int i = 0; i < 8; ++i)
                qh[(size_t)(mrow + i) * RDIM + ncol] = (_Float16)acc[rr][cb][i];
        }
    }
}

// ---- Kernel 2: landmark-scores GEMM (per n,h): 64x128 tile, K=64 -----------
// scores = qh(f16) @ lmh(f16, x0.125 folded)^T, f32 out
__global__ void __launch_bounds__(256)
scores_gemm_kernel(const _Float16* __restrict__ qh, const _Float16* __restrict__ lmh,
                   float* __restrict__ scores) {
    __shared__ _Float16 As2[64][72];
    __shared__ _Float16 Bs2[128][72];

    const int t     = threadIdx.x;
    const int cBase = blockIdx.x * 64;
    const int h     = blockIdx.y;
    const int n     = blockIdx.z;

    // stage A: 64x64 halves = 512 x 16B chunks (async)
    #pragma unroll
    for (int i = 0; i < 2; ++i) {
        int e = t + i * 256;
        int m = e >> 3, kc = (e & 7) * 8;
        async_copy_b128(&As2[m][kc],
                        qh + (size_t)(n * CC + cBase + m) * RDIM + h * PH + kc);
    }
    // stage B: 128x64 halves = 1024 x 16B chunks (async)
    #pragma unroll
    for (int i = 0; i < 4; ++i) {
        int e = t + i * 256;
        int d = e >> 3, kc = (e & 7) * 8;
        async_copy_b128(&Bs2[d][kc],
                        lmh + (((size_t)n * DCH + d) * NH + h) * PH + kc);
    }
    wait_async_0();
    __syncthreads();

    const int lane   = t & 31;
    const int wv     = t >> 5;
    const int half   = lane >> 4;
    const int l16    = lane & 15;
    const int rband  = wv & 3;
    const int cgroup = (wv >> 2) * 4;

    v8f acc[4] = { {}, {}, {}, {} };
    #pragma unroll
    for (int kt = 0; kt < 64; kt += 32) {
        v16h a;
        const int m = rband * 16 + l16;
        #pragma unroll
        for (int v = 0; v < 8; ++v) {
            int kb = kt + kA(v, half);
            a[2 * v]     = As2[m][kb];
            a[2 * v + 1] = As2[m][kb + 1];
        }
        #pragma unroll
        for (int cb = 0; cb < 4; ++cb) {
            v16h b;
            const int d = (cgroup + cb) * 16 + l16;
            #pragma unroll
            for (int r = 0; r < 8; ++r) {
                int kb = kt + kA(r, half);
                b[2 * r]     = Bs2[d][kb];
                b[2 * r + 1] = Bs2[d][kb + 1];
            }
            acc[cb] = __builtin_amdgcn_wmma_f32_16x16x32_f16(
                false, a, false, b, (short)0, acc[cb], false, false);
        }
    }

    #pragma unroll
    for (int cb = 0; cb < 4; ++cb) {
        const int d = (cgroup + cb) * 16 + l16;
        #pragma unroll
        for (int i = 0; i < 8; ++i) {
            const int m = rband * 16 + half * 8 + i;
            scores[(((size_t)n * CC + cBase + m) * NH + h) * DCH + d] = acc[cb][i];
        }
    }
}

// ---- Kernel 3: top-16 + index sort + softplus cumsum (one wave per row) ----
__global__ void __launch_bounds__(256)
topk_weights_kernel(const float* __restrict__ scores,
                    float* __restrict__ out_w, int* __restrict__ out_i) {
    const int gtid = blockIdx.x * blockDim.x + threadIdx.x;
    const int wid  = gtid >> 5;          // (n*C+c)*H+h
    const int lane = threadIdx.x & 31;
    const int c    = (wid >> 3) & (CC - 1);

    const float* srow = scores + (size_t)wid * DCH;
    const int nvis = c >> 6;

    float s[4];
    #pragma unroll
    for (int j = 0; j < 4; ++j) {
        int d = lane + 32 * j;
        s[j] = (d < nvis) ? srow[d] : neg_inf();
    }

    int used = 0, selIdx = 0;
    #pragma unroll
    for (int it = 0; it < TOPK; ++it) {
        float bs = neg_inf();
        int   bi = 0x7fffffff;
        #pragma unroll
        for (int j = 0; j < 4; ++j) {
            if (!(used & (1 << j))) {
                int d = lane + 32 * j;
                if (s[j] > bs || (s[j] == bs && d < bi)) { bs = s[j]; bi = d; }
            }
        }
        #pragma unroll
        for (int off = 16; off; off >>= 1) {
            float os = __shfl_xor(bs, off, 32);
            int   oi = __shfl_xor(bi, off, 32);
            if (os > bs || (os == bs && oi < bi)) { bs = os; bi = oi; }
        }
        if (lane == it) selIdx = bi;
        if ((bi & 31) == lane) used |= 1 << ((bi >> 5) & 3);
    }

    int val = (lane < TOPK && lane < nvis) ? selIdx : -1;

    int rank = 0;
    #pragma unroll
    for (int i = 0; i < TOPK; ++i) {
        int vi = __shfl(val, i, 32);
        if (vi > val || (vi == val && i < lane)) ++rank;
    }
    int sval = -1;
    #pragma unroll
    for (int i = 0; i < TOPK; ++i) {
        int vi = __shfl(val, i, 32);
        int ri = __shfl(rank, i, 32);
        if (ri == lane) sval = vi;
    }

    const bool inv  = (sval < 0);
    const int  idxf = inv ? 0 : sval;
    float sc = inv ? neg_inf() : srow[idxf];

    float sp = (sc > 15.0f) ? sc : log1pf(__expf(fminf(sc, 15.0f)));

    float cum = sp;
    #pragma unroll
    for (int off = 1; off < TOPK; off <<= 1) {
        float tv = __shfl_up(cum, off, 16);
        if ((lane & 15) >= off) cum += tv;
    }

    if (lane < TOPK) {
        out_w[(size_t)wid * TOPK + lane] = __expf(sc - cum);
        out_i[(size_t)wid * TOPK + lane] = idxf;
    }
}

// ---- launch ----------------------------------------------------------------
extern "C" void kernel_launch(void* const* d_in, const int* in_sizes, int n_in,
                              void* d_out, int out_size, void* d_ws, size_t ws_size,
                              hipStream_t stream) {
    const float* hs  = (const float*)d_in[0];  // (N,C,HID)
    const float* lm  = (const float*)d_in[1];  // (N,D,H,64)
    const float* pnw = (const float*)d_in[2];  // (HID,)
    const float* W   = (const float*)d_in[3];  // (RDIM,HID)

    char* ws = (char*)d_ws;
    _Float16* xh     = (_Float16*)ws;                       // 16384*4096 f16 (128 MB)
    ws += (size_t)ROWS * HID * sizeof(_Float16);
    _Float16* qh     = (_Float16*)ws;                       // 16384*512 f16  (16 MB)
    ws += (size_t)ROWS * RDIM * sizeof(_Float16);
    float*    scores = (float*)ws;                          // 16384*8*128 f32 (64 MB)
    ws += (size_t)ROWS * NH * DCH * sizeof(float);
    _Float16* wf     = (_Float16*)ws;                       // 512*4096 f16   (4 MB)
    ws += (size_t)RDIM * HID * sizeof(_Float16);
    _Float16* lmh    = (_Float16*)ws;                       // 2*128*8*64 f16 (256 KB)

    float* out_w = (float*)d_out;
    int*   out_i = (int*)d_out + (size_t)ROWS * NH * TOPK;

    rms_norm_f16_kernel<<<ROWS / 8, 256, 0, stream>>>(hs, pnw, xh);
    f32_to_f16_kernel<<<(RDIM * HID / 4) / 256, 256, 0, stream>>>(W, wf, 1.0f);
    f32_to_f16_kernel<<<(NN * DCH * NH * PH / 4) / 256, 256, 0, stream>>>(lm, lmh, 0.125f);

    dim3 g1(ROWS / 128, RDIM / 128);
    proj_gemm_kernel<<<g1, 256, 0, stream>>>(xh, wf, qh);

    dim3 g2(CC / 64, NH, NN);
    scores_gemm_kernel<<<g2, 256, 0, stream>>>(qh, lmh, scores);

    topk_weights_kernel<<<(ROWS * NH) / 8, 256, 0, stream>>>(scores, out_w, out_i);
}
